// LSTM_36902359007346
// MI455X (gfx1250) — matrix operand order repaired and checked by reading, not compiled, not run on hardware
//
#include <hip/hip_runtime.h>
#include <hip/hip_bf16.h>
#include <stdint.h>

// Problem dims
#define B_  64
#define T_  2048
#define D_  512
#define H_  512
#define NWG 32            // one WG per 16-column H chunk
#define NTHR 256          // 8 wave32 per WG

// LDS A-matrix row stride in halves (1024 + 8 pad to break bank alignment)
#define AROW 1032

// Async global->LDS availability (gfx1250 ASYNCcnt path)
#if defined(__has_builtin)
#  if __has_builtin(__builtin_amdgcn_global_load_async_to_lds_b128)
#    define ASYNC_MODE 1
#  else
#    define ASYNC_MODE 0
#  endif
#else
#  define ASYNC_MODE 0
#endif

typedef __attribute__((ext_vector_type(16))) __bf16 v16bf;
typedef __attribute__((ext_vector_type(8)))  float  v8f;
typedef __attribute__((ext_vector_type(4)))  int    v4i;
typedef __attribute__((address_space(1))) v4i gv4i;   // global-AS int4
typedef __attribute__((address_space(3))) v4i lv4i;   // LDS-AS int4

union FragU { uint4 u[2]; v16bf v; };

__device__ __forceinline__ unsigned short f32_bf16(float f) {
  unsigned u = __float_as_uint(f);
  u += 0x7FFFu + ((u >> 16) & 1u);          // round-to-nearest-even
  return (unsigned short)(u >> 16);
}
__device__ __forceinline__ float sigmoid_f(float x) { return 1.0f / (1.0f + __expf(-x)); }
__device__ __forceinline__ float tanh_f(float x)    { return 1.0f - 2.0f / (__expf(2.0f * x) + 1.0f); }

// ---------------------------------------------------------------------------
// Pack [Wi;Wh] (1024 x 2048 fp32) into bf16 WMMA B-operand fragment order.
// Fragment (nt, kt) = 32 lanes x 16 halves (1 KB). Lane n<16: col n, K=k0+{0..7,16..23};
// lane n>=16: col n-16, K=k0+{8..15,24..31}.   nt = gate*32 + hchunk.
// ---------------------------------------------------------------------------
__global__ void pack_w_kernel(const float* wii, const float* wif, const float* wig, const float* wio,
                              const float* whi, const float* whf, const float* whg, const float* who,
                              unsigned short* __restrict__ Wp) {
  int p = blockIdx.x * 256 + threadIdx.x;          // < 1024*2048 = 2,097,152
  int frag = p >> 9;                               // /512
  int r    = p & 511;
  int lane = r >> 4;
  int slot = r & 15;
  int kt = frag & 31;
  int nt = frag >> 5;
  int g   = nt >> 5;                               // gate 0..3 (i,f,g,o)
  int hch = nt & 31;                               // H chunk 0..31
  int col = hch * 16 + (lane & 15);                // 0..511
  int K   = kt * 32 + ((slot & 8) << 1) + ((lane >> 4) << 3) + (slot & 7);  // 0..1023
  const float* W;
  int kk;
  if (K < 512) {
    W  = (g == 0) ? wii : (g == 1) ? wif : (g == 2) ? wig : wio;
    kk = K;
  } else {
    W  = (g == 0) ? whi : (g == 1) ? whf : (g == 2) ? whg : who;
    kk = K - 512;
  }
  Wp[p] = f32_bf16(W[kk * H_ + col]);
}

__global__ void pack_bias_kernel(const float* bii, const float* bif, const float* big, const float* bio,
                                 const float* bhi, const float* bhf, const float* bhg, const float* bho,
                                 float* __restrict__ bias) {
  int idx = blockIdx.x * 256 + threadIdx.x;        // < 2048
  int g = idx >> 9, j = idx & 511;
  const float* bi = (g == 0) ? bii : (g == 1) ? bif : (g == 2) ? big : bio;
  const float* bh = (g == 0) ? bhi : (g == 1) ? bhf : (g == 2) ? bhg : bho;
  bias[idx] = bi[j] + bh[j];
}

// ---------------------------------------------------------------------------
// Device-wide barrier: monotonic counter, release add + acquire spin.
// ---------------------------------------------------------------------------
__device__ __forceinline__ void grid_sync(unsigned* cnt, unsigned target) {
  __threadfence();
  __syncthreads();
  if (threadIdx.x == 0) {
    __hip_atomic_fetch_add(cnt, 1u, __ATOMIC_ACQ_REL, __HIP_MEMORY_SCOPE_AGENT);
    while (__hip_atomic_load(cnt, __ATOMIC_ACQUIRE, __HIP_MEMORY_SCOPE_AGENT) < target) {
      __builtin_amdgcn_s_sleep(2);
    }
  }
  __syncthreads();
}

// ---------------------------------------------------------------------------
// Persistent LSTM kernel: 32 WGs x 256 threads, one step per grid barrier.
// ---------------------------------------------------------------------------
__global__ void __launch_bounds__(NTHR)
lstm_persistent(const float* __restrict__ x,
                const unsigned short* __restrict__ Wp,
                const float* __restrict__ bias,
                unsigned short* __restrict__ h0buf,
                unsigned short* __restrict__ h1buf,
                unsigned* __restrict__ counter,
                float* __restrict__ out) {
  // 320 KB LDS per WGP on CDNA5: 129 KB A-staging + 16 KB gate buffer.
  __shared__ unsigned short Alds[B_ * AROW];       // 132,096 B
  __shared__ float gbuf[4][B_][16];                // 16,384 B

  const int hch  = blockIdx.x;                     // H chunk: cols [hch*16, hch*16+16)
  const int tid  = threadIdx.x;
  const int wave = tid >> 5;                       // 0..7
  const int lane = tid & 31;
  const int g    = wave & 3;                       // gate
  const int mt0  = (wave >> 2) * 2;                // M-tile pair {mt0, mt0+1}

  // per-wave constant B-fragment base (packed, 1KB per fragment, 32B per lane)
  const unsigned short* Bb =
      Wp + (((size_t)(g * 32 + hch) * 32) << 9) + (lane << 4);

  // per-lane A-fragment addressing per ISA 16-bit A layout:
  // lane<16: row = l,    K offsets {0..7, 16..23}
  // lane>=16: row = l-16, K offsets {8..15, 24..31}
  const int arow_lo = (lane & 15);
  const int koff    = (lane < 16) ? 0 : 8;

  // staging assignment: thread -> (batch row, 128-col quarter)
  const int srow = tid >> 2;                       // 0..63
  const int sq   = tid & 3;                        // 0..3
  const float* xbase = x + ((size_t)srow * T_) * D_ + sq * 128;
  unsigned short* adst  = &Alds[srow * AROW + sq * 128];
  unsigned short* ahdst = &Alds[srow * AROW + D_ + sq * 128];

  // epilogue constants: per-thread element set, hoisted bias, c-state in VGPRs
  int   em[4], eidx[4];
  float eb[4][4];
  float creg[4];
#pragma unroll
  for (int q = 0; q < 4; ++q) {
    int e  = tid + q * 256;                        // 0..1023
    int m  = e >> 4;
    int cc = e & 15;
    int hcol = hch * 16 + cc;
    em[q]   = m;
    eidx[q] = m * H_ + hcol;
    eb[q][0] = bias[hcol];
    eb[q][1] = bias[512 + hcol];
    eb[q][2] = bias[1024 + hcol];
    eb[q][3] = bias[1536 + hcol];
    creg[q] = 0.0f;
  }

  for (int t = 0; t < T_; ++t) {
    const unsigned short* hin  = (t & 1) ? h1buf : h0buf;
    unsigned short*       hout = (t & 1) ? h0buf : h1buf;

    // ---- stage A = [x_t | h_t] into LDS as bf16 ----
    {
      const unsigned short* hsrc = hin + srow * H_ + sq * 128;
#if ASYNC_MODE == 1
      // issue async DMA h->LDS first (ASYNCcnt), overlap with x conversion below
#pragma unroll
      for (int j = 0; j < 16; ++j) {
        __builtin_amdgcn_global_load_async_to_lds_b128(
            (gv4i*)(hsrc + j * 8), (lv4i*)(ahdst + j * 8), 0, 0);
      }
#else
      {
        const uint4* hs4 = (const uint4*)hsrc;
        uint4* hd4 = (uint4*)ahdst;
#pragma unroll 4
        for (int j = 0; j < 16; ++j) hd4[j] = hs4[j];
      }
#endif

      const float* xrow = xbase + (size_t)t * D_;
#pragma unroll 4
      for (int k = 0; k < 128; k += 4) {
        float4 f = *(const float4*)(xrow + k);
        uint2 pk;
        pk.x = (unsigned)f32_bf16(f.x) | ((unsigned)f32_bf16(f.y) << 16);
        pk.y = (unsigned)f32_bf16(f.z) | ((unsigned)f32_bf16(f.w) << 16);
        *(uint2*)(adst + k) = pk;
      }
      // prefetch next timestep's x row into cache (global_prefetch_b8)
      if (t + 1 < T_) __builtin_prefetch(xrow + D_, 0, 0);

#if ASYNC_MODE == 1
      asm volatile("s_wait_asynccnt 0x0" ::: "memory");
#endif
    }
    __syncthreads();

    // ---- GEMM: 2 tiles x 32 K-steps of v_wmma_f32_16x16x32_bf16 ----
    v8f acc0 = {};
    v8f acc1 = {};
#pragma unroll 4
    for (int kt = 0; kt < 32; ++kt) {
      const unsigned short* bp = Bb + (kt << 9);
      FragU bf;
      bf.u[0] = *(const uint4*)(bp);
      bf.u[1] = *(const uint4*)(bp + 8);

      const unsigned short* a0p = &Alds[(mt0 * 16 + arow_lo) * AROW + kt * 32 + koff];
      const unsigned short* a1p = &Alds[((mt0 + 1) * 16 + arow_lo) * AROW + kt * 32 + koff];
      FragU a0, a1;
      a0.u[0] = *(const uint4*)(a0p);
      a0.u[1] = *(const uint4*)(a0p + 16);
      a1.u[0] = *(const uint4*)(a1p);
      a1.u[1] = *(const uint4*)(a1p + 16);

      acc0 = __builtin_amdgcn_wmma_f32_16x16x32_bf16(false, a0.v, false, bf.v,
                                                     (short)0, acc0, false, false);
      acc1 = __builtin_amdgcn_wmma_f32_16x16x32_bf16(false, a1.v, false, bf.v,
                                                     (short)0, acc1, false, false);
    }

    // ---- scatter accumulators to gate buffer (C/D layout: VGPR r, lane) ----
    {
      int colw = lane & 15;
      int rb = (lane < 16) ? 0 : 8;
#pragma unroll
      for (int r = 0; r < 8; ++r) {
        gbuf[g][mt0 * 16 + rb + r][colw]        = acc0[r];
        gbuf[g][(mt0 + 1) * 16 + rb + r][colw]  = acc1[r];
      }
    }
    __syncthreads();

    // ---- elementwise gate math, c/h update (4 elems/thread, c in VGPRs) ----
#pragma unroll
    for (int q = 0; q < 4; ++q) {
      int m  = em[q];
      int cc = (tid + q * 256) & 15;
      float gi = gbuf[0][m][cc] + eb[q][0];
      float gf = gbuf[1][m][cc] + eb[q][1];
      float gg = gbuf[2][m][cc] + eb[q][2];
      float go = gbuf[3][m][cc] + eb[q][3];
      float it = sigmoid_f(gi);
      float ft = sigmoid_f(gf);
      float gt = tanh_f(gg);
      float ot = sigmoid_f(go);
      float cn = ft * creg[q] + it * gt;
      float hn = ot * tanh_f(cn);
      creg[q] = cn;
      hout[eidx[q]] = f32_bf16(hn);
      if (t == T_ - 1) {                           // final outputs (h_T, c_T)
        out[eidx[q]]           = hn;
        out[B_ * H_ + eidx[q]] = cn;
      }
    }

    grid_sync(counter, (unsigned)(t + 1) * NWG);
  }
}

// ---------------------------------------------------------------------------
// Workspace layout (bytes):
//   0        : packed W bf16   (4,194,304)
//   4194304  : fused bias f32  (8,192)
//   4202496  : h buf 0 bf16    (65,536)  \
//   4268032  : h buf 1 bf16    (65,536)   > zeroed every launch
//   4333568  : barrier counter (64)      /
// ---------------------------------------------------------------------------
extern "C" void kernel_launch(void* const* d_in, const int* in_sizes, int n_in,
                              void* d_out, int out_size, void* d_ws, size_t ws_size,
                              hipStream_t stream) {
  (void)in_sizes; (void)n_in; (void)out_size; (void)ws_size;

  const float* x   = (const float*)d_in[0];
  const float* wii = (const float*)d_in[1];
  const float* wif = (const float*)d_in[2];
  const float* wig = (const float*)d_in[3];
  const float* wio = (const float*)d_in[4];
  const float* whi = (const float*)d_in[5];
  const float* whf = (const float*)d_in[6];
  const float* whg = (const float*)d_in[7];
  const float* who = (const float*)d_in[8];
  const float* bii = (const float*)d_in[9];
  const float* bif = (const float*)d_in[10];
  const float* big = (const float*)d_in[11];
  const float* bio = (const float*)d_in[12];
  const float* bhi = (const float*)d_in[13];
  const float* bhf = (const float*)d_in[14];
  const float* bhg = (const float*)d_in[15];
  const float* bho = (const float*)d_in[16];

  uint8_t* ws = (uint8_t*)d_ws;
  unsigned short* Wp   = (unsigned short*)(ws);
  float*          bias = (float*)(ws + 4194304);
  unsigned short* h0   = (unsigned short*)(ws + 4202496);
  unsigned short* h1   = (unsigned short*)(ws + 4268032);
  unsigned*       cnt  = (unsigned*)(ws + 4333568);

  // zero h0/h1/counter every launch (graph-capture safe memset node)
  (void)hipMemsetAsync(ws + 4202496, 0, 131136, stream);

  pack_w_kernel<<<(1024 * 2048) / 256, 256, 0, stream>>>(
      wii, wif, wig, wio, whi, whf, whg, who, Wp);
  pack_bias_kernel<<<2048 / 256, 256, 0, stream>>>(
      bii, bif, big, bio, bhi, bhf, bhg, bho, bias);

  lstm_persistent<<<NWG, NTHR, 0, stream>>>(
      x, Wp, bias, h0, h1, cnt, (float*)d_out);
}
